// LstmEncoder_52913997087095
// MI455X (gfx1250) — compile-verified, matrix operand order checked by reading, not run on hardware
//
#include <hip/hip_runtime.h>
#include <hip/hip_bf16.h>

// ---------------------------------------------------------------------------
// LSTM encoder for MI455X (gfx1250, wave32, WMMA).
//   B=32, S=1024, D=UNITS=512, gates N=2048, LAYERS=2.
// Strategy:
//   * bf16 v_wmma_f32_16x16x32_bf16 everywhere (fp32 accumulate)
//   * xp = X@W + b as a parallel WMMA GEMM, 32x64 tile/wave, K double-buffered
//   * recurrence: ONE persistent 1024-thread workgroup; h/c in 160 KB LDS;
//     U stays L2-resident; B fragments shared across M tiles + pipelined
//   * xp stream marked non-temporal so it does not evict U/W/X0 from L2
// ---------------------------------------------------------------------------

typedef __attribute__((ext_vector_type(16))) __bf16 v16bf;
typedef __attribute__((ext_vector_type(8)))  __bf16 v8bf;   // 16 bytes
typedef __attribute__((ext_vector_type(8)))  float  v8f;

static constexpr int Bb    = 32;    // batch
static constexpr int Ss    = 1024;  // seq len
static constexpr int Dd    = 512;   // embed dim == units == K
static constexpr int UNITS = 512;
static constexpr int NG    = 4 * UNITS; // 2048 gate columns
static constexpr int Kk    = 512;

union Frag { v16bf v; v8bf h[2]; };

// Load one 16x32 bf16 WMMA fragment (A or B) from a row-major tile with
// leading dim `ld`. CDNA5 16-bit A/B layout:
//   lanes 0-15 : row lr, K = {0..7} then {16..23}
//   lanes 16-31: row lr, K = {8..15} then {24..31}
// i.e. two contiguous 16-byte chunks per lane.
__device__ __forceinline__ v16bf load_frag(const __bf16* tile, int ld, int lane) {
    const int half = lane >> 4, lr = lane & 15;
    const __bf16* q = tile + (size_t)lr * ld + half * 8;
    Frag f;
    f.h[0] = *(const v8bf*)(q);
    f.h[1] = *(const v8bf*)(q + 16);
    return f.v;
}

__device__ __forceinline__ v8f wmma_bf16(v16bf a, v16bf b, v8f c) {
    return __builtin_amdgcn_wmma_f32_16x16x32_bf16(
        /*neg_a=*/false, a, /*neg_b=*/false, b,
        /*c_mod=*/(short)0, c, /*reuse_a=*/false, /*reuse_b=*/false);
}

__device__ __forceinline__ float sigm(float x) { return 1.0f / (1.0f + __expf(-x)); }

// ---------------------------------------------------------------------------
// Pack W and U into bf16, transposed to N-major [layer][n=2048][k=512] so B
// fragments are contiguous 16B loads per lane.
// ---------------------------------------------------------------------------
__global__ void pack_wu(const float* __restrict__ W, const float* __restrict__ U,
                        __bf16* __restrict__ Wt, __bf16* __restrict__ Ut) {
    const int n_per = 2 * NG * Kk;                  // per tensor
    int idx = blockIdx.x * blockDim.x + threadIdx.x;
    if (idx >= 2 * n_per) return;
    int which = idx / n_per;
    int r = idx - which * n_per;
    int k  = r & (Kk - 1);
    int t2 = r >> 9;            // / 512
    int n  = t2 & (NG - 1);
    int l  = t2 >> 11;          // / 2048
    const float* src = which ? U : W;
    __bf16* dst = which ? Ut : Wt;
    dst[r] = (__bf16)src[((size_t)l * Kk + k) * NG + n];
}

// ---------------------------------------------------------------------------
// Embedding gather into time-major bf16 A-matrix: X0[t][b][d] = emb[x[b][t]][d]
// ---------------------------------------------------------------------------
__global__ void gather_emb(const int* __restrict__ x, const float* __restrict__ emb,
                           __bf16* __restrict__ X0) {
    int idx = blockIdx.x * blockDim.x + threadIdx.x;   // over 32768*512
    int d = idx & (Dd - 1);
    int r = idx >> 9;           // row = t*32 + b
    int t = r >> 5;
    int b = r & 31;
    int tok = x[b * Ss + t];
    X0[(size_t)r * Dd + d] = (__bf16)emb[(size_t)tok * Dd + d];
}

// ---------------------------------------------------------------------------
// xp = A[32768,512](bf16) @ Bt^T + bias -> C[32768,2048] fp32 (non-temporal).
// 8 waves/block; each wave computes a 32x64 strip: acc[2 mTiles][4 nTiles].
// K loop software-pipelined (double-buffered fragments): 8 WMMA per 12 loads.
// ---------------------------------------------------------------------------
__global__ __launch_bounds__(256) void gemm_xp(const __bf16* __restrict__ A,
                                               const __bf16* __restrict__ Bt,
                                               const float* __restrict__ bias,
                                               float* __restrict__ C) {
    const int lane = threadIdx.x & 31;
    const int wave = threadIdx.x >> 5;
    const int lr = lane & 15, half = lane >> 4;
    const int task   = blockIdx.x * 8 + wave;   // 1024 mGroups * 32 nGroups
    const int mGroup = task >> 5;
    const int nGroup = task & 31;
    const __bf16* aBase = A + (size_t)mGroup * 32 * Kk;
    const __bf16* bBase = Bt + (size_t)nGroup * 64 * Kk;

    v8f acc[2][4];
#pragma unroll
    for (int j = 0; j < 4; ++j) {
        float bv = bias[nGroup * 64 + j * 16 + lr];
#pragma unroll
        for (int e = 0; e < 8; ++e) { acc[0][j][e] = bv; acc[1][j][e] = bv; }
    }

    v16bf aC[2], bC[4], aN[2], bN[4];
#pragma unroll
    for (int mt = 0; mt < 2; ++mt) aC[mt] = load_frag(aBase + mt * 16 * Kk, Kk, lane);
#pragma unroll
    for (int j = 0; j < 4; ++j)    bC[j]  = load_frag(bBase + (size_t)j * 16 * Kk, Kk, lane);

#pragma unroll
    for (int kb = 0; kb < Kk; kb += 32) {
        if (kb + 32 < Kk) {            // prefetch next K-slice while WMMAs run
#pragma unroll
            for (int mt = 0; mt < 2; ++mt)
                aN[mt] = load_frag(aBase + mt * 16 * Kk + kb + 32, Kk, lane);
#pragma unroll
            for (int j = 0; j < 4; ++j)
                bN[j] = load_frag(bBase + (size_t)j * 16 * Kk + kb + 32, Kk, lane);
        }
#pragma unroll
        for (int j = 0; j < 4; ++j) {
            acc[0][j] = wmma_bf16(aC[0], bC[j], acc[0][j]);
            acc[1][j] = wmma_bf16(aC[1], bC[j], acc[1][j]);
        }
        if (kb + 32 < Kk) {
#pragma unroll
            for (int mt = 0; mt < 2; ++mt) aC[mt] = aN[mt];
#pragma unroll
            for (int j = 0; j < 4; ++j)    bC[j]  = bN[j];
        }
    }

    // C/D layout: col = nBase+lr, rows mBase + half*8 + v. Stream out NT.
#pragma unroll
    for (int mt = 0; mt < 2; ++mt)
#pragma unroll
        for (int j = 0; j < 4; ++j) {
            float* cp = C + ((size_t)mGroup * 32 + mt * 16 + half * 8) * NG
                          + nGroup * 64 + j * 16 + lr;
#pragma unroll
            for (int v = 0; v < 8; ++v)
                __builtin_nontemporal_store(acc[mt][j][v], cp + (size_t)v * NG);
        }
}

// ---------------------------------------------------------------------------
// Persistent single-workgroup LSTM recurrence for one layer.
// 1024 threads = 32 waves. Wave w owns unit-columns [16w,16w+16) of all four
// gates for both 16-row M tiles -> gate math is wave-local.
// kb-outer K loop: each U fragment loaded once, shared by both M tiles, and
// double-buffered so WMMAs overlap the L2 fetch of the next K slice.
// h (fp32 + bf16 shadow) and c (fp32) live in LDS (160 KB).
// ---------------------------------------------------------------------------
__global__ __launch_bounds__(1024, 1) void lstm_recurrence(
        const float* __restrict__ xp,    // [S*32][2048] fp32 (x-projection + bias)
        const __bf16* __restrict__ Ut,   // this layer's U, N-major [2048][512]
        const int* __restrict__ x,       // [32][1024] tokens (mask = x==0)
        __bf16* __restrict__ hseq,       // out: [S*32][512] bf16, or null
        float* __restrict__ outHC) {     // out: h_last|c_last (2*32*512 fp32), or null
    __shared__ __bf16 sHbf[Bb * UNITS];  // 32 KB
    __shared__ float  sH  [Bb * UNITS];  // 64 KB
    __shared__ float  sC  [Bb * UNITS];  // 64 KB

    const int tid = threadIdx.x;
    for (int i = tid; i < Bb * UNITS; i += 1024) {
        sH[i] = 0.0f; sC[i] = 0.0f; sHbf[i] = (__bf16)0.0f;
    }
    __syncthreads();

    const int lane = tid & 31, wave = tid >> 5;
    const int lr = lane & 15, half = lane >> 4;
    // This wave's 4 gate-column rows in Ut (N-major): n = g*512 + wave*16 + lr
    const __bf16* uBase[4];
#pragma unroll
    for (int g = 0; g < 4; ++g)
        uBase[g] = Ut + (size_t)(g * UNITS + wave * 16) * UNITS;

    for (int t = 0; t < Ss; ++t) {
        v8f acc[2][4];
        // init accumulators from xp (z_in = xW + b); NT: streamed once
#pragma unroll
        for (int mt = 0; mt < 2; ++mt)
#pragma unroll
            for (int g = 0; g < 4; ++g) {
                const float* xpp = xp + ((size_t)t * Bb + mt * 16 + half * 8) * NG
                                      + g * UNITS + wave * 16 + lr;
#pragma unroll
                for (int v = 0; v < 8; ++v)
                    acc[mt][g][v] = __builtin_nontemporal_load(xpp + (size_t)v * NG);
            }

        // z += h @ U : A from LDS (ds_load_b128), B from L2-resident Ut.
        v16bf bC[4], bN[4];
#pragma unroll
        for (int g = 0; g < 4; ++g) bC[g] = load_frag(uBase[g], UNITS, lane);
#pragma unroll
        for (int kb = 0; kb < UNITS; kb += 32) {
            if (kb + 32 < UNITS) {     // prefetch next K slice of U
#pragma unroll
                for (int g = 0; g < 4; ++g)
                    bN[g] = load_frag(uBase[g] + kb + 32, UNITS, lane);
            }
            v16bf a0 = load_frag(&sHbf[kb], UNITS, lane);
            v16bf a1 = load_frag(&sHbf[16 * UNITS + kb], UNITS, lane);
#pragma unroll
            for (int g = 0; g < 4; ++g) {
                acc[0][g] = wmma_bf16(a0, bC[g], acc[0][g]);
                acc[1][g] = wmma_bf16(a1, bC[g], acc[1][g]);
            }
            if (kb + 32 < UNITS) {
#pragma unroll
                for (int g = 0; g < 4; ++g) bC[g] = bN[g];
            }
        }
        __syncthreads();   // all reads of sHbf for step t complete

        // gates -> c/h update (wave-local; each (brow,col) touched by one lane)
#pragma unroll
        for (int mt = 0; mt < 2; ++mt) {
#pragma unroll
            for (int v = 0; v < 8; ++v) {
                int brow = mt * 16 + half * 8 + v;
                int col  = wave * 16 + lr;
                bool msk = (x[brow * Ss + t] == 0);  // mask True => compute step
                float zi = acc[mt][0][v], zf = acc[mt][1][v];
                float zg = acc[mt][2][v], zo = acc[mt][3][v];
                int idx = brow * UNITS + col;
                float cold = sC[idx], hold = sH[idx];
                float cn = sigm(zf) * cold + sigm(zi) * tanhf(zg);
                float hn = sigm(zo) * tanhf(cn);
                float c2 = msk ? cn : cold;
                float h2 = msk ? hn : hold;
                sC[idx] = c2; sH[idx] = h2; sHbf[idx] = (__bf16)h2;
                if (hseq)
                    hseq[((size_t)t * Bb + brow) * UNITS + col] = (__bf16)h2;
            }
        }
        __syncthreads();   // new h visible before step t+1 reads
    }

    if (outHC) {
        for (int i = tid; i < Bb * UNITS; i += 1024) {
            outHC[i] = sH[i];                 // h_last
            outHC[Bb * UNITS + i] = sC[i];    // c_last
        }
    }
}

// ---------------------------------------------------------------------------
extern "C" void kernel_launch(void* const* d_in, const int* in_sizes, int n_in,
                              void* d_out, int out_size, void* d_ws, size_t ws_size,
                              hipStream_t stream) {
    (void)in_sizes; (void)n_in; (void)out_size; (void)ws_size;
    const int*   x   = (const int*)d_in[0];     // [32][1024]
    const float* emb = (const float*)d_in[1];   // [32000][512]
    const float* W   = (const float*)d_in[2];   // [2][512][2048]
    const float* U   = (const float*)d_in[3];   // [2][512][2048]
    const float* b   = (const float*)d_in[4];   // [2][2048]

    char* ws = (char*)d_ws;
    size_t o = 0;
    float*  xp = (float*)(ws + o);  o += (size_t)Bb * Ss * NG * sizeof(float);   // 268 MB
    __bf16* X0 = (__bf16*)(ws + o); o += (size_t)Bb * Ss * Dd * sizeof(__bf16);  // 32 MB
    __bf16* h1 = (__bf16*)(ws + o); o += (size_t)Bb * Ss * UNITS * sizeof(__bf16);
    __bf16* Wt = (__bf16*)(ws + o); o += (size_t)2 * NG * Kk * sizeof(__bf16);
    __bf16* Ut = (__bf16*)(ws + o);

    pack_wu   <<<(2 * 2 * NG * Kk) / 256, 256, 0, stream>>>(W, U, Wt, Ut);
    gather_emb<<<(Bb * Ss * Dd) / 256,    256, 0, stream>>>(x, emb, X0);

    const int gemmBlocks = (1024 * 32) / 8;   // mGroups * nGroups / wavesPerBlock
    // layer 0
    gemm_xp<<<gemmBlocks, 256, 0, stream>>>(X0, Wt, b, xp);
    lstm_recurrence<<<1, 1024, 0, stream>>>(xp, Ut, x, h1, nullptr);
    // layer 1
    gemm_xp<<<gemmBlocks, 256, 0, stream>>>(h1, Wt + (size_t)NG * Kk, b + NG, xp);
    lstm_recurrence<<<1, 1024, 0, stream>>>(xp, Ut + (size_t)NG * Kk, x, nullptr,
                                            (float*)d_out);
}
